// SetLevelCrossAttentionPrototype_55310588838256
// MI455X (gfx1250) — matrix operand order repaired and checked by reading, not compiled, not run on hardware
//
#include <hip/hip_runtime.h>
#include <hip/hip_bf16.h>
#include <math.h>
#include <stdint.h>

// Problem constants (match reference)
#define B_  64
#define L_  2048
#define D_  128
#define H_  4
#define A_  128
#define M_  16
#define V_  100000
#define HD_ 32
#define GAMMA_ 0.3f

typedef __attribute__((ext_vector_type(16))) __bf16        v16bf;
typedef __attribute__((ext_vector_type(8)))  float         v8f;
typedef __attribute__((ext_vector_type(2)))  float         v2f;
typedef __attribute__((ext_vector_type(4)))  unsigned int  v4u;
typedef __attribute__((ext_vector_type(8)))  int           v8i;
typedef __attribute__((ext_vector_type(4)))  int           v4i;

// workspace layout (float offsets)
#define WS_PHIP   0        // 16*128   = 2048
#define WS_SIZEP  2048     // 16
#define WS_PHIQ   2064     // 64*128   = 8192
#define WS_SUMPS  10256    // 64*16    = 1024
#define WS_Y      11280    // 64*4*128 = 32768
#define WS_ZERON  2064     // floats to zero (atomic accumulators only)

__device__ __forceinline__ float sigmoidf_(float x) { return 1.0f / (1.0f + __expf(-x)); }

// ---------------------------------------------------------------------------
// TDM descriptor builders (CDNA5 ISA ch.8: Tensor DMA Descriptor D#)
// Group 0: [1:0]=count=1 (user mode), [63:32]=lds_addr, [120:64]=global_addr,
//          [127:126]=type=2 ("image").
// Group 1: [15:0]=workgroup_mask=0, [17:16]=data_size (2 -> 4 bytes),
//          [79:48]=tensor_dim0, [111:80]=tensor_dim1, [127:112]=tile_dim0,
//          [143:128]=tile_dim1, [159:144]=tile_dim2=0,
//          [207:160]=tensor_dim0_stride, [255:208]=tensor_dim1_stride=0.
// Groups 2/3 (and the trailing v8i on this toolchain's 6-arg builtin) are
// zero: 2-D tensor, no iteration/gather.
// ---------------------------------------------------------------------------
__device__ __forceinline__ v4u tdm_g0(const void* gaddr, unsigned lds_off) {
    const unsigned long long ga = (unsigned long long)(uintptr_t)gaddr;
    v4u g0;
    g0[0] = 1u;                                                // count=1
    g0[1] = lds_off;                                           // lds_addr (bytes)
    g0[2] = (unsigned)(ga & 0xFFFFFFFFu);                      // global_addr lo
    g0[3] = (unsigned)((ga >> 32) & 0x01FFFFFFu) | (2u << 30); // addr hi | type=2
    return g0;
}
__device__ __forceinline__ v8i tdm_g1(unsigned dim0, unsigned dim1,
                                      unsigned tile0, unsigned tile1,
                                      unsigned long long stride0) {
    v8i g1;
    g1[0] = 0x00020000;                                        // mask=0, 4-byte elems
    g1[1] = (int)((dim0 & 0xFFFFu) << 16);                     // abar=0 | dim0 lo16
    g1[2] = (int)(((dim0 >> 16) & 0xFFFFu) | ((dim1 & 0xFFFFu) << 16));
    g1[3] = (int)(((dim1 >> 16) & 0xFFFFu) | ((tile0 & 0xFFFFu) << 16));
    g1[4] = (int)(tile1 & 0xFFFFu);                            // tile1 | tile2=0
    g1[5] = (int)(stride0 & 0xFFFFFFFFull);                    // stride0 lo
    g1[6] = (int)((stride0 >> 32) & 0xFFFFull);                // stride0 hi | s1 lo=0
    g1[7] = 0;
    return g1;
}
__device__ __forceinline__ unsigned lds_off_(const void* p) {
    // LDS aperture flat address: low 32 bits are the LDS byte offset
    return (unsigned)(unsigned long long)(uintptr_t)p;
}

// ---------------------------------------------------------------------------
// Kernel 0: zero the atomic accumulator regions (PhiP, sizeP)
// ---------------------------------------------------------------------------
__global__ void k_zero(float* __restrict__ ws) {
    for (int i = threadIdx.x; i < WS_ZERON; i += blockDim.x) ws[i] = 0.0f;
}

// ---------------------------------------------------------------------------
// Kernel 1: PhiP = sigmoid(phi_logits) @ atom_emb   (M=16, N=128, K=100000)
//           sizeP = row sums of sigmoid(phi_logits)
// Double-buffered TDM pipeline: wave 0 issues two tensor_load_to_lds per
// 32-K chunk (A: 16x32 strided tile of phi_logits; B: 32x128 tile of
// atom_emb), waits with s_wait_tensorcnt(2) so the next chunk's DMAs stay
// in flight behind the current WMMA. Wave w computes N-tile w; partials
// committed with global_atomic_add_f32.
// ---------------------------------------------------------------------------
__global__ void __launch_bounds__(256) k_phiP(const float* __restrict__ phi_logits,
                                              const float* __restrict__ atom_emb,
                                              float* __restrict__ ws) {
    __shared__ float Ab[2][M_ * 32];     // 2 x 2KB : sigmoid-input logits tiles
    __shared__ float Bb[2][32 * A_];     // 2 x 16KB: atom_emb K-chunk tiles

    float* PhiP  = ws + WS_PHIP;
    float* sizeP = ws + WS_SIZEP;

    const int tid  = threadIdx.x;
    const int lane = tid & 31;
    const int w    = tid >> 5;           // wave id == N-tile index (0..7)
    const int row  = lane & 15;
    const int col  = lane & 15;
    const int kA   = (lane < 16) ? 0 : 8;    // A operand K base (ISA §7.12.2)
    const int kB   = (lane < 16) ? 0 : 8;    // B operand K base
    const int nchunks = V_ / 32;             // 3125 exactly
    const int step = gridDim.x;

    v8f acc;
    #pragma unroll
    for (int e = 0; e < 8; ++e) acc[e] = 0.0f;
    float szacc = 0.0f;

    v4i z4; z4[0] = 0; z4[1] = 0; z4[2] = 0; z4[3] = 0;
    v8i z8;
    #pragma unroll
    for (int e = 0; e < 8; ++e) z8[e] = 0;

    int c = blockIdx.x;
    // prologue: stage first chunk into buffer 0 (one DMA pair, issued by wave 0)
    if (c < nchunks && tid < 32) {
        const int k0 = c * 32;
        v4u g0a = tdm_g0(phi_logits + k0, lds_off_(&Ab[0][0]));
        v8i g1a = tdm_g1(V_ - k0, M_, 32, M_, V_);          // 16 rows x 32 cols, stride V
        __builtin_amdgcn_tensor_load_to_lds(g0a, g1a, z4, z4, z8, 0);
        v4u g0b = tdm_g0(atom_emb + (size_t)k0 * A_, lds_off_(&Bb[0][0]));
        v8i g1b = tdm_g1(A_, V_ - k0, A_, 32, A_);          // 32 rows x 128 cols
        __builtin_amdgcn_tensor_load_to_lds(g0b, g1b, z4, z4, z8, 0);
    }

    int bufI = 0;
    for (; c < nchunks; c += step) {
        const int cn = c + step;
        if (tid < 32) {
            if (cn < nchunks) {
                const int k0n = cn * 32;
                v4u g0a = tdm_g0(phi_logits + k0n, lds_off_(&Ab[bufI ^ 1][0]));
                v8i g1a = tdm_g1(V_ - k0n, M_, 32, M_, V_);
                __builtin_amdgcn_tensor_load_to_lds(g0a, g1a, z4, z4, z8, 0);
                v4u g0b = tdm_g0(atom_emb + (size_t)k0n * A_, lds_off_(&Bb[bufI ^ 1][0]));
                v8i g1b = tdm_g1(A_, V_ - k0n, A_, 32, A_);
                __builtin_amdgcn_tensor_load_to_lds(g0b, g1b, z4, z4, z8, 0);
                __builtin_amdgcn_s_wait_tensorcnt((short)2);  // current chunk landed
            } else {
                __builtin_amdgcn_s_wait_tensorcnt((short)0);  // drain on last chunk
            }
        }
        __syncthreads();

        const float* Abuf = Ab[bufI];
        const float* Bbuf = Bb[bufI];
        v16bf a;
        #pragma unroll
        for (int e = 0; e < 16; ++e) {
            const int kk = kA + (e & 7) + ((e >> 3) << 4);
            const float pv = sigmoidf_(Abuf[row * 32 + kk]);
            szacc += pv;
            a[e] = (__bf16)pv;
        }
        v16bf b;
        #pragma unroll
        for (int e = 0; e < 16; ++e)
            b[e] = (__bf16)Bbuf[(kB + (e & 7) + ((e >> 3) << 4)) * A_ + w * 16 + col];
        acc = __builtin_amdgcn_wmma_f32_16x16x32_bf16(
            false, a, false, b, (short)0, acc, false, false);

        __syncthreads();
        bufI ^= 1;
    }

    // commit: C/D layout -> lane holds (M = e + 8*(lane>=16), N = w*16+col)
    const int mOff = (lane < 16) ? 0 : 8;
    #pragma unroll
    for (int e = 0; e < 8; ++e)
        atomicAdd(&PhiP[(e + mOff) * A_ + w * 16 + col], acc[e]);
    if (w == 0) atomicAdd(&sizeP[row], szacc);  // lanes L / L+16 cover disjoint K
}

// ---------------------------------------------------------------------------
// Kernel 2: CSR gather-reduce. One block per batch b (128 threads).
// PhiQ[b][d]   = sum_{i in [offs[b],offs[b+1])} atom_emb[ids[i]][d]
// sumpS[b][m]  = sum_i sigmoid(phi_logits[m][ids[i]])
// Register accumulation (no atomics); global_prefetch_b8 on upcoming rows.
// ---------------------------------------------------------------------------
__global__ void k_phiQ(const float* __restrict__ atom_emb,
                       const float* __restrict__ phi_logits,
                       const int* __restrict__ query_ids,
                       const int* __restrict__ query_offs,
                       float* __restrict__ ws) {
    float* PhiQ  = ws + WS_PHIQ;
    float* sumpS = ws + WS_SUMPS;
    const int b   = blockIdx.x;
    const int tid = threadIdx.x;   // 128
    const int beg = query_offs[b];
    const int end = query_offs[b + 1];

    float accq = 0.0f, accp = 0.0f;
    for (int i = beg; i < end; ++i) {
        const int id = query_ids[i];
        if (tid < 4 && i + 8 < end) {
            const int idn = query_ids[i + 8];
            __builtin_prefetch(&atom_emb[(size_t)idn * A_ + tid * 32], 0, 1);
        }
        accq += atom_emb[(size_t)id * A_ + tid];        // 512B coalesced
        if (tid < M_) accp += sigmoidf_(phi_logits[(size_t)tid * V_ + id]);
    }
    PhiQ[b * A_ + tid] = accq;
    if (tid < M_) sumpS[b * M_ + tid] = accp;
}

// ---------------------------------------------------------------------------
// Kernel 3: all the small B x M stuff, single block of 128 threads.
// AQ=PhiQ@W_A.T, BP=PhiP@W_B.T, scores -> softmax -> attn, Vp, Z=attn@Vp,
// Y[b][h][d] = sum_j Z[b][j] * W_out[d][h*32+j]   (rank-4 fold of W_out)
// ---------------------------------------------------------------------------
__global__ void k_small(const int* __restrict__ query_offs,
                        const float* __restrict__ W_A,
                        const float* __restrict__ W_B,
                        const float* __restrict__ W_val,
                        const float* __restrict__ b_val,
                        const float* __restrict__ W_out,
                        const float* __restrict__ size_w,
                        float* __restrict__ ws) {
    __shared__ float AQ[B_][A_];     // 32 KB
    __shared__ float BP[M_][A_];     //  8 KB
    __shared__ float SC[B_][M_];     //  4 KB  (scores, then attn)
    __shared__ float VP[M_][HD_];    //  2 KB
    __shared__ float ZS[B_][HD_];    //  8 KB
    const float* PhiP  = ws + WS_PHIP;
    const float* sizeP = ws + WS_SIZEP;
    const float* PhiQ  = ws + WS_PHIQ;
    const float* sumpS = ws + WS_SUMPS;
    float* Y = ws + WS_Y;
    const int tid = threadIdx.x;     // 128
    const float sw0 = size_w[0], sw1 = size_w[1];

    for (int idx = tid; idx < M_ * A_; idx += 128) {
        const int m = idx / A_, a = idx % A_;
        float s = 0.0f;
        for (int k = 0; k < A_; ++k) s += PhiP[m * A_ + k] * W_B[a * A_ + k];
        BP[m][a] = s;
    }
    for (int idx = tid; idx < B_ * A_; idx += 128) {
        const int b = idx / A_, a = idx % A_;
        float s = 0.0f;
        for (int k = 0; k < A_; ++k) s += PhiQ[b * A_ + k] * W_A[a * A_ + k];
        AQ[b][a] = s;
    }
    __syncthreads();

    for (int idx = tid; idx < B_ * M_; idx += 128) {
        const int b = idx / M_, m = idx % M_;
        float d = 0.0f;
        for (int k = 0; k < A_; ++k) d += AQ[b][k] * BP[m][k];
        const float sq = (float)(query_offs[b + 1] - query_offs[b]);
        const float sp = sizeP[m];
        const float delta = sq + sp - 2.0f * sumpS[b * M_ + m];
        SC[b][m] = -GAMMA_ * delta + d + sw0 * sq + sw1 * sp;   // BETA=TAU=1
    }
    __syncthreads();

    if (tid < B_) {   // softmax over M=16 per row
        float mx = -1e30f;
        for (int m = 0; m < M_; ++m) mx = fmaxf(mx, SC[tid][m]);
        float s = 0.0f;
        for (int m = 0; m < M_; ++m) { const float e = __expf(SC[tid][m] - mx); SC[tid][m] = e; s += e; }
        const float inv = 1.0f / s;
        for (int m = 0; m < M_; ++m) SC[tid][m] *= inv;
    }
    for (int idx = tid; idx < M_ * HD_; idx += 128) {
        const int m = idx / HD_, j = idx % HD_;
        float s = b_val[j];
        for (int k = 0; k < A_; ++k) s += PhiP[m * A_ + k] * W_val[j * A_ + k];
        VP[m][j] = s;
    }
    __syncthreads();

    for (int idx = tid; idx < B_ * HD_; idx += 128) {
        const int b = idx / HD_, j = idx % HD_;
        float s = 0.0f;
        for (int m = 0; m < M_; ++m) s += SC[b][m] * VP[m][j];
        ZS[b][j] = s;
    }
    __syncthreads();

    for (int idx = tid; idx < B_ * H_ * A_; idx += 128) {
        const int b = idx >> 9;
        const int h = (idx >> 7) & 3;
        const int d = idx & 127;
        float s = 0.0f;
        for (int j = 0; j < HD_; ++j) s += ZS[b][j] * W_out[d * D_ + h * HD_ + j];
        Y[idx] = s;
    }
}

// ---------------------------------------------------------------------------
// Kernel 4: the big memory-bound pass. 1024 blocks x 256 threads; each wave
// owns 16 consecutive tokens. Phase 1: gate logits via float4 loads +
// wave32 __shfl_xor butterflies, softmax over H=4 -> LDS. Phase 2: per
// 16-token tile, out = gates(16x4) @ Y[b](4x128) + b_out via 8x
// V_WMMA_F32_16X16X4_F32 (A/B f32, K=4 native; no padding, full precision).
// ---------------------------------------------------------------------------
__global__ void k_out(const float* __restrict__ token_states,
                      const float* __restrict__ W_gate,
                      const float* __restrict__ b_gate,
                      const float* __restrict__ b_out,
                      const float* __restrict__ ws,
                      float* __restrict__ out) {
    __shared__ float Ys[H_][D_];
    __shared__ float bo[D_];
    __shared__ float gs[128][H_];
    const float* Y = ws + WS_Y;

    const int batch = blockIdx.x >> 4;          // 16 blocks per batch
    const int chunk = blockIdx.x & 15;
    const long tokBase = (long)batch * L_ + (long)chunk * 128;   // row in (B*L)
    const int tid  = threadIdx.x;               // 256
    const int lane = tid & 31;
    const int w    = tid >> 5;                  // wave 0..7

    for (int i = tid; i < H_ * D_; i += 256) Ys[i >> 7][i & 127] = Y[batch * (H_ * D_) + i];
    if (tid < D_) bo[tid] = b_out[tid];

    float4 wg[H_];
    #pragma unroll
    for (int h = 0; h < H_; ++h) wg[h] = ((const float4*)(W_gate + h * D_))[lane];
    float bg[H_];
    #pragma unroll
    for (int h = 0; h < H_; ++h) bg[h] = b_gate[h];

    // ---- phase 1: gates for this wave's 16 tokens ----
    for (int j = 0; j < 16; ++j) {
        const long tok = tokBase + w * 16 + j;
        if (j + 2 < 16 && lane < 4)
            __builtin_prefetch(token_states + (tok + 2) * D_ + lane * 32, 0, 1);
        const float4 ts = ((const float4*)(token_states + tok * D_))[lane];  // 512B/wave
        float p[H_];
        #pragma unroll
        for (int h = 0; h < H_; ++h)
            p[h] = ts.x * wg[h].x + ts.y * wg[h].y + ts.z * wg[h].z + ts.w * wg[h].w;
        #pragma unroll
        for (int m = 16; m >= 1; m >>= 1)
            #pragma unroll
            for (int h = 0; h < H_; ++h)
                p[h] += __shfl_xor(p[h], m, 32);
        float mx = p[0] + bg[0];
        #pragma unroll
        for (int h = 1; h < H_; ++h) mx = fmaxf(mx, p[h] + bg[h]);
        float e[H_], s = 0.0f;
        #pragma unroll
        for (int h = 0; h < H_; ++h) { e[h] = __expf(p[h] + bg[h] - mx); s += e[h]; }
        const float inv = 1.0f / s;
        if (lane == 0)
            #pragma unroll
            for (int h = 0; h < H_; ++h) gs[w * 16 + j][h] = e[h] * inv;
    }
    __syncthreads();   // also guarantees full EXEC reconvergence before WMMA

    // ---- phase 2: 16x128 output tile via f32 WMMA (K=4) ----
    // A 16x4 f32 layout: lane L<16 -> row L, K {0,1}; lane L+16 -> row L, K {2,3}
    const int col = lane & 15;
    const int kb  = (lane < 16) ? 0 : 2;
    v2f a2;
    a2[0] = gs[w * 16 + col][kb];
    a2[1] = gs[w * 16 + col][kb + 1];
    const int mOff = (lane < 16) ? 0 : 8;
    #pragma unroll
    for (int t = 0; t < 8; ++t) {
        const int n = t * 16 + col;
        v2f b2;
        b2[0] = Ys[kb][n];
        b2[1] = Ys[kb + 1][n];
        v8f c;
        #pragma unroll
        for (int e2 = 0; e2 < 8; ++e2) c[e2] = bo[n];   // bias broadcast down rows
        v8f d = __builtin_amdgcn_wmma_f32_16x16x4_f32(
            false, a2, false, b2, (short)0, c, false, false);
        #pragma unroll
        for (int e2 = 0; e2 < 8; ++e2) {
            const long tok = tokBase + w * 16 + e2 + mOff;
            out[tok * D_ + n] = d[e2];    // 64B coalesced per half-wave
        }
    }
}

// ---------------------------------------------------------------------------
extern "C" void kernel_launch(void* const* d_in, const int* in_sizes, int n_in,
                              void* d_out, int out_size, void* d_ws, size_t ws_size,
                              hipStream_t stream) {
    const float* token_states = (const float*)d_in[0];
    const int*   query_ids    = (const int*)d_in[1];
    const int*   query_offs   = (const int*)d_in[2];
    const float* atom_emb     = (const float*)d_in[3];
    const float* phi_logits   = (const float*)d_in[4];
    const float* W_A          = (const float*)d_in[5];
    const float* W_B          = (const float*)d_in[6];
    const float* W_val        = (const float*)d_in[7];
    const float* b_val        = (const float*)d_in[8];
    const float* W_gate       = (const float*)d_in[9];
    const float* b_gate       = (const float*)d_in[10];
    const float* W_out        = (const float*)d_in[11];
    const float* b_out        = (const float*)d_in[12];
    const float* size_w       = (const float*)d_in[13];
    float* ws  = (float*)d_ws;
    float* out = (float*)d_out;

    k_zero<<<1, 256, 0, stream>>>(ws);
    k_phiP<<<16, 256, 0, stream>>>(phi_logits, atom_emb, ws);   // TDM-fed WMMA
    k_phiQ<<<B_, 128, 0, stream>>>(atom_emb, phi_logits, query_ids, query_offs, ws);
    k_small<<<1, 128, 0, stream>>>(query_offs, W_A, W_B, W_val, b_val, W_out, size_w, ws);
    k_out<<<(B_ * L_) / 128, 256, 0, stream>>>(token_states, W_gate, b_gate, b_out, ws, out);
}
